// TransformerBlockCausal_75874892251198
// MI455X (gfx1250) — matrix operand order repaired and checked by reading, not compiled, not run on hardware
//
#include <hip/hip_runtime.h>
#include <hip/hip_bf16.h>

// ---------------------------------------------------------------------------
// Transformer block (pre-LN, RoPE, causal attention, MLP) for MI455X gfx1250.
// B=2 T=2048 E=1024 H=16 D=64 FF=4096.
// GEMMs: 128x128 block tiles, 4 waves, 64x64 register blocking per wave,
// double-buffered LDS staging via global_load_async_to_lds_b128 (ASYNCcnt),
// compute via v_wmma_f32_16x16x32_bf16.
// Attention: flash-style, 32 queries per wave, K/V fragments reused across
// two Q sub-tiles.
// ---------------------------------------------------------------------------

typedef __bf16 bf16_t;
typedef __attribute__((ext_vector_type(16))) __bf16 v16bf;
typedef __attribute__((ext_vector_type(8)))  __bf16 v8bf;
typedef __attribute__((ext_vector_type(8)))  float  v8f;

constexpr int cB  = 2;
constexpr int cT  = 2048;
constexpr int cE  = 1024;
constexpr int cH  = 16;
constexpr int cD  = 64;
constexpr int cFF = 4096;
constexpr int cM  = cB * cT;          // 4096 rows of activations

#define FLAG_RELU    1
#define FLAG_OUTBF16 2

// Async global->LDS copy of 16 bytes, per-lane addresses on both sides.
__device__ __forceinline__ void async_g2l_b128(unsigned int lds_addr,
                                               const void* gaddr) {
  asm volatile("global_load_async_to_lds_b128 %0, %1, off"
               :: "v"(lds_addr), "v"((unsigned long long)(uintptr_t)gaddr)
               : "memory");
}
__device__ __forceinline__ void wait_async0() {
  asm volatile("s_wait_asynccnt 0x0" ::: "memory");
}

// ---------------------------------------------------------------------------
// Convert weight W[K,N] (f32, row-major) into bf16 WMMA-B-fragment order:
// dest[((n0/16)*(K/32) + k0/32)*512 + lane*16 + i]  holds
//   W[k0 + (lane>>4)*16 + i][n0 + (lane&15)]
// ---------------------------------------------------------------------------
__global__ void pack_w_kernel(const float* __restrict__ W, bf16_t* __restrict__ Wp,
                              int K, int N) {
  size_t d = (size_t)blockIdx.x * blockDim.x + threadIdx.x;
  if (d >= (size_t)K * N) return;
  int i    = (int)(d & 15);
  int lane = (int)((d >> 4) & 31);
  size_t t = d >> 9;
  int kTiles = K >> 5;
  int k0 = (int)(t % kTiles) << 5;
  int n0 = (int)(t / kTiles) << 4;
  int k = k0 + ((lane >> 4) << 4) + i;
  int n = n0 + (lane & 15);
  Wp[d] = (bf16_t)W[(size_t)k * N + n];
}

// ---------------------------------------------------------------------------
// Row LayerNorm: y_bf16 = (x - mu) * rsqrt(var+eps) * g + b.  One block / row.
// ---------------------------------------------------------------------------
__global__ __launch_bounds__(256)
void ln_kernel(const float* __restrict__ x, const float* __restrict__ g,
               const float* __restrict__ be, bf16_t* __restrict__ y) {
  int row = blockIdx.x;
  int tid = threadIdx.x;
  const float* xr = x + (size_t)row * cE;
  float s = 0.f, ss = 0.f;
  for (int j = tid; j < cE; j += 256) { float v = xr[j]; s += v; ss += v * v; }
  for (int off = 16; off > 0; off >>= 1) {
    s  += __shfl_down(s,  off);
    ss += __shfl_down(ss, off);
  }
  __shared__ float sh[18];
  int w = tid >> 5;
  if ((tid & 31) == 0) { sh[w] = s; sh[8 + w] = ss; }
  __syncthreads();
  if (tid == 0) {
    float a = 0.f, b2 = 0.f;
    for (int i = 0; i < 8; ++i) { a += sh[i]; b2 += sh[8 + i]; }
    float mu  = a / cE;
    float var = b2 / cE - mu * mu;
    sh[16] = mu;
    sh[17] = rsqrtf(var + 1e-5f);
  }
  __syncthreads();
  float mu = sh[16], rs = sh[17];
  bf16_t* yr = y + (size_t)row * cE;
  for (int j = tid; j < cE; j += 256)
    yr[j] = (bf16_t)(((xr[j] - mu) * rs) * g[j] + be[j]);
}

// ---------------------------------------------------------------------------
// WMMA GEMM, 128x128 output tile per 128-thread block (4 waves, 2x2).
// Each wave: 64x64 in registers = 4x4 tiles of 16x16 f32 (128 VGPRs).
// Double-buffered LDS: slab kt+1's async group is issued right after the
// barrier, overlapping DMA with the 16 WMMAs of slab kt.
// ---------------------------------------------------------------------------
__global__ __launch_bounds__(128)
void gemm_wmma_kernel(const bf16_t* __restrict__ A, const bf16_t* __restrict__ Bp,
                      const float* __restrict__ bias, const float* __restrict__ resid,
                      void* __restrict__ C, int M, int N, int K, int flags) {
  __shared__ __align__(32) bf16_t As[2][8 * 32 * 16];  // frag order per buffer
  __shared__ __align__(32) bf16_t Bs[2][8 * 512];

  const int tid  = threadIdx.x;
  const int lane = tid & 31;
  const int w    = tid >> 5;          // wave id 0..3
  const int wr   = w >> 1, wc = w & 1;
  const int half = lane >> 4, lr = lane & 15;
  const int tileM = blockIdx.y << 7;
  const int tileN = blockIdx.x << 7;
  const int kTiles = K >> 5;

  unsigned int asBase[2], bsBase[2];
  asBase[0] = (unsigned int)(uintptr_t)(void*)As[0];
  asBase[1] = (unsigned int)(uintptr_t)(void*)As[1];
  bsBase[0] = (unsigned int)(uintptr_t)(void*)Bs[0];
  bsBase[1] = (unsigned int)(uintptr_t)(void*)Bs[1];

  // --- A staging plan: 512 16B-chunks, 4 per thread.
  // chunk c: msub=c>>6, r=(c>>2)&15, kg=c&3  (kg selects k-elements 8*kg..)
  // LDS frag slot: lane' = r + (kg>>1)*16, element offset (kg&1)*8.
  int aRow[4]; int aKoff[4]; unsigned int aLds[4];
#pragma unroll
  for (int j = 0; j < 4; ++j) {
    int c = tid * 4 + j;
    int msub = c >> 6, r = (c >> 2) & 15, kg = c & 3;
    aRow[j]  = tileM + msub * 16 + r;
    aKoff[j] = kg * 8;
    aLds[j]  = (unsigned int)(((msub * 32 + (r + (kg >> 1) * 16)) * 16 +
                               (kg & 1) * 8) * 2);
  }
  // --- B staging plan: thread copies 64 contiguous bytes of one nsub panel.
  const int bNsub = tid >> 4, bInner = tid & 15;
  const size_t bPanel = ((size_t)(blockIdx.x * 8 + bNsub) * kTiles) * 512;
  const unsigned int bLds = (unsigned int)((bNsub * 512 + bInner * 32) * 2);

  v8f acc[4][4];
#pragma unroll
  for (int mi = 0; mi < 4; ++mi)
#pragma unroll
    for (int ni = 0; ni < 4; ++ni)
#pragma unroll
      for (int i = 0; i < 8; ++i) acc[mi][ni][i] = 0.f;

  // prologue: slab 0 -> buffer 0
  {
    const int k0 = 0;
#pragma unroll
    for (int j = 0; j < 4; ++j)
      async_g2l_b128(asBase[0] + aLds[j], A + (size_t)aRow[j] * K + k0 + aKoff[j]);
    const bf16_t* bsrc = Bp + bPanel + bInner * 32;
#pragma unroll
    for (int j = 0; j < 4; ++j)
      async_g2l_b128(bsBase[0] + bLds + j * 16, bsrc + j * 8);
  }

  for (int kt = 0; kt < kTiles; ++kt) {
    const int cur = kt & 1;
    wait_async0();                   // slab kt landed (only outstanding group)
    __syncthreads();                 // all waves landed it; prev reads done
    if (kt + 1 < kTiles) {           // overlap DMA of slab kt+1 with compute
      const int nxt = (kt + 1) & 1;
      const int k0 = (kt + 1) << 5;
#pragma unroll
      for (int j = 0; j < 4; ++j)
        async_g2l_b128(asBase[nxt] + aLds[j],
                       A + (size_t)aRow[j] * K + k0 + aKoff[j]);
      const bf16_t* bsrc = Bp + bPanel + (size_t)(kt + 1) * 512 + bInner * 32;
#pragma unroll
      for (int j = 0; j < 4; ++j)
        async_g2l_b128(bsBase[nxt] + bLds + j * 16, bsrc + j * 8);
    }

    v16bf af[4], bfv[4];
#pragma unroll
    for (int mi = 0; mi < 4; ++mi)
      af[mi] = *(const v16bf*)(As[cur] + ((wr * 4 + mi) * 32 + lane) * 16);
#pragma unroll
    for (int ni = 0; ni < 4; ++ni)
      bfv[ni] = *(const v16bf*)(Bs[cur] + (wc * 4 + ni) * 512 + lane * 16);
#pragma unroll
    for (int mi = 0; mi < 4; ++mi)
#pragma unroll
      for (int ni = 0; ni < 4; ++ni)
        acc[mi][ni] = __builtin_amdgcn_wmma_f32_16x16x32_bf16(
            false, af[mi], false, bfv[ni], (short)0, acc[mi][ni], false, false);
  }

  // epilogue
#pragma unroll
  for (int mi = 0; mi < 4; ++mi) {
#pragma unroll
    for (int ni = 0; ni < 4; ++ni) {
      int colBase = tileN + (wc * 4 + ni) * 16 + lr;
      float bv = bias[colBase];
#pragma unroll
      for (int i = 0; i < 8; ++i) {
        int r = tileM + (wr * 4 + mi) * 16 + half * 8 + i;
        float v = acc[mi][ni][i] + bv;
        if (flags & FLAG_RELU) v = fmaxf(v, 0.f);
        if (resid) v += resid[(size_t)r * N + colBase];
        if (flags & FLAG_OUTBF16)
          ((bf16_t*)C)[(size_t)r * N + colBase] = (bf16_t)v;
        else
          ((float*)C)[(size_t)r * N + colBase] = v;
      }
    }
  }
}

// ---------------------------------------------------------------------------
// RoPE + repack: q,k f32 [B*T,E] -> bf16 [bh,T,64] with rotation applied;
// v f32 -> bf16 transposed [bh,64,T] so P@V B-fragments are contiguous.
// ---------------------------------------------------------------------------
__global__ __launch_bounds__(256)
void rope_pack_kernel(const float* __restrict__ qf, const float* __restrict__ kf,
                      const float* __restrict__ vf,
                      bf16_t* __restrict__ qh, bf16_t* __restrict__ kh,
                      bf16_t* __restrict__ vt) {
  size_t idx = (size_t)blockIdx.x * 256 + threadIdx.x;   // over B*H*T*D
  int d  = (int)(idx & 63);
  int t  = (int)((idx >> 6) & (cT - 1));
  int bh = (int)(idx >> 17);
  int h = bh & (cH - 1), b = bh >> 4;
  size_t src = ((size_t)(b * cT + t)) * cE + h * cD + d;

  float ang = (float)t * powf(10000.0f, -(float)(2 * (d >> 1)) / (float)cD);
  float sn = __sinf(ang), cs = __cosf(ang);

  float qv = qf[src];
  float qr = (d & 1) ? qf[src - 1] : -qf[src + 1];
  float kv = kf[src];
  float kr = (d & 1) ? kf[src - 1] : -kf[src + 1];

  size_t dstq = ((size_t)bh * cT + t) * cD + d;
  qh[dstq] = (bf16_t)(qv * cs + qr * sn);
  kh[dstq] = (bf16_t)(kv * cs + kr * sn);
  vt[((size_t)bh * cD + d) * cT + t] = (bf16_t)vf[src];
}

// ---------------------------------------------------------------------------
// Flash attention, one wave per (bh, 32-query tile); streams 32-key chunks.
// K/V fragments loaded once per chunk and reused for both 16-query sub-tiles:
// 8 score WMMAs + 8 P@V WMMAs per chunk.  Online softmax via shfl_xor within
// 16-lane groups; P goes C-layout -> LDS -> A-layout.
// ---------------------------------------------------------------------------
__global__ __launch_bounds__(32)
void attn_kernel(const bf16_t* __restrict__ Q, const bf16_t* __restrict__ Km,
                 const bf16_t* __restrict__ Vt, bf16_t* __restrict__ Z) {
  int lane = threadIdx.x;
  int half = lane >> 4, lr = lane & 15;
  int qBase = blockIdx.x << 5;          // 32 queries per wave
  int bh = blockIdx.y;
  int b = bh >> 4, h = bh & (cH - 1);

  const bf16_t* Qh = Q  + (size_t)bh * cT * cD;
  const bf16_t* Kh = Km + (size_t)bh * cT * cD;
  const bf16_t* Vh = Vt + (size_t)bh * cD * cT;

  __shared__ __align__(32) bf16_t pshm[32 * 32];

  // Q A-fragments [qsub][dstep], kept in registers for the whole loop
  v16bf aQ[2][2];
#pragma unroll
  for (int qs = 0; qs < 2; ++qs) {
    const bf16_t* qrow = Qh + (size_t)(qBase + qs * 16 + lr) * cD + half * 16;
#pragma unroll
    for (int dstep = 0; dstep < 2; ++dstep) {
      v8bf x0 = *(const v8bf*)(qrow + dstep * 32);
      v8bf x1 = *(const v8bf*)(qrow + dstep * 32 + 8);
#pragma unroll
      for (int i = 0; i < 8; ++i) {
        aQ[qs][dstep][i] = x0[i];
        aQ[qs][dstep][8 + i] = x1[i];
      }
    }
  }

  float m[2][8], l[2][8];
  v8f accO[2][4];
#pragma unroll
  for (int qs = 0; qs < 2; ++qs) {
#pragma unroll
    for (int i = 0; i < 8; ++i) { m[qs][i] = -1e30f; l[qs][i] = 0.f; }
#pragma unroll
    for (int t = 0; t < 4; ++t)
#pragma unroll
      for (int i = 0; i < 8; ++i) accO[qs][t][i] = 0.f;
  }

  const float scale = 0.125f;   // 1/sqrt(64)

  for (int kc = 0; kc < qBase + 32; kc += 32) {
    v8f s[2][2];
#pragma unroll
    for (int qs = 0; qs < 2; ++qs)
#pragma unroll
      for (int t = 0; t < 2; ++t)
#pragma unroll
        for (int i = 0; i < 8; ++i) s[qs][t][i] = 0.f;

#pragma unroll
    for (int dstep = 0; dstep < 2; ++dstep) {
      int d0 = dstep * 32;
      v16bf b0 = *(const v16bf*)(Kh + (size_t)(kc + lr) * cD + d0 + half * 16);
      v16bf b1 = *(const v16bf*)(Kh + (size_t)(kc + 16 + lr) * cD + d0 + half * 16);
#pragma unroll
      for (int qs = 0; qs < 2; ++qs) {
        s[qs][0] = __builtin_amdgcn_wmma_f32_16x16x32_bf16(
            false, aQ[qs][dstep], false, b0, (short)0, s[qs][0], false, false);
        s[qs][1] = __builtin_amdgcn_wmma_f32_16x16x32_bf16(
            false, aQ[qs][dstep], false, b1, (short)0, s[qs][1], false, false);
      }
    }

#pragma unroll
    for (int qs = 0; qs < 2; ++qs) {
#pragma unroll
      for (int i = 0; i < 8; ++i) {
        int r = qBase + qs * 16 + half * 8 + i;
        float v0 = s[qs][0][i] * scale;
        float v1 = s[qs][1][i] * scale;
        if (kc + lr      > r) v0 = -1e30f;
        if (kc + 16 + lr > r) v1 = -1e30f;
        float mx = fmaxf(v0, v1);
        for (int off = 8; off > 0; off >>= 1) mx = fmaxf(mx, __shfl_xor(mx, off));
        float mn = fmaxf(m[qs][i], mx);
        float alpha = __expf(m[qs][i] - mn);
        float p0 = __expf(v0 - mn);
        float p1 = __expf(v1 - mn);
        float rs = p0 + p1;
        for (int off = 8; off > 0; off >>= 1) rs += __shfl_xor(rs, off);
        l[qs][i] = l[qs][i] * alpha + rs;
        m[qs][i] = mn;
#pragma unroll
        for (int t = 0; t < 4; ++t) accO[qs][t][i] *= alpha;
        int rl = qs * 16 + half * 8 + i;
        pshm[rl * 32 + lr]      = (bf16_t)p0;
        pshm[rl * 32 + 16 + lr] = (bf16_t)p1;
      }
    }
    __syncthreads();

    v16bf aP[2];
#pragma unroll
    for (int qs = 0; qs < 2; ++qs) {
      const bf16_t* prow = pshm + (qs * 16 + lr) * 32 + half * 16;
      v8bf p0v = *(const v8bf*)(prow);
      v8bf p1v = *(const v8bf*)(prow + 8);
#pragma unroll
      for (int i = 0; i < 8; ++i) { aP[qs][i] = p0v[i]; aP[qs][8 + i] = p1v[i]; }
    }

#pragma unroll
    for (int t = 0; t < 4; ++t) {
      v16bf bv = *(const v16bf*)(Vh + (size_t)(t * 16 + lr) * cT + kc + half * 16);
#pragma unroll
      for (int qs = 0; qs < 2; ++qs)
        accO[qs][t] = __builtin_amdgcn_wmma_f32_16x16x32_bf16(
            false, aP[qs], false, bv, (short)0, accO[qs][t], false, false);
    }
    __syncthreads();
  }

#pragma unroll
  for (int qs = 0; qs < 2; ++qs) {
#pragma unroll
    for (int t = 0; t < 4; ++t) {
#pragma unroll
      for (int i = 0; i < 8; ++i) {
        int r = qBase + qs * 16 + half * 8 + i;
        float o = accO[qs][t][i] / l[qs][i];
        Z[((size_t)(b * cT + r)) * cE + h * cD + t * 16 + lr] = (bf16_t)o;
      }
    }
  }
}

// ---------------------------------------------------------------------------
// Host-side orchestration
// ---------------------------------------------------------------------------
extern "C" void kernel_launch(void* const* d_in, const int* in_sizes, int n_in,
                              void* d_out, int out_size, void* d_ws, size_t ws_size,
                              hipStream_t stream) {
  const float* x     = (const float*)d_in[0];
  const float* Wq    = (const float*)d_in[1];
  const float* bq    = (const float*)d_in[2];
  const float* Wk    = (const float*)d_in[3];
  const float* bk    = (const float*)d_in[4];
  const float* Wv    = (const float*)d_in[5];
  const float* bv    = (const float*)d_in[6];
  const float* Wo    = (const float*)d_in[7];
  const float* bo    = (const float*)d_in[8];
  const float* g1    = (const float*)d_in[9];
  const float* beta1 = (const float*)d_in[10];
  const float* g2    = (const float*)d_in[11];
  const float* beta2 = (const float*)d_in[12];
  const float* W1    = (const float*)d_in[13];
  const float* b1    = (const float*)d_in[14];
  const float* W2    = (const float*)d_in[15];
  const float* b2    = (const float*)d_in[16];

  char* ws = (char*)d_ws;
  const size_t MB = 1ull << 20;
  bf16_t* Wqp = (bf16_t*)(ws + 0 * MB);
  bf16_t* Wkp = (bf16_t*)(ws + 2 * MB);
  bf16_t* Wvp = (bf16_t*)(ws + 4 * MB);
  bf16_t* Wop = (bf16_t*)(ws + 6 * MB);
  bf16_t* W1p = (bf16_t*)(ws + 8 * MB);
  bf16_t* W2p = (bf16_t*)(ws + 16 * MB);
  bf16_t* ybf = (bf16_t*)(ws + 24 * MB);   // LN1 / LN2 output (reused)
  float*  qf  = (float*) (ws + 32 * MB);
  float*  kf  = (float*) (ws + 48 * MB);
  float*  vf  = (float*) (ws + 64 * MB);
  float*  x1  = (float*) (ws + 32 * MB);   // reuses qf after rope_pack
  bf16_t* hid = (bf16_t*)(ws + 48 * MB);   // reuses kf/vf after rope_pack
  bf16_t* qh  = (bf16_t*)(ws + 80 * MB);
  bf16_t* kh  = (bf16_t*)(ws + 88 * MB);
  bf16_t* vt  = (bf16_t*)(ws + 96 * MB);
  bf16_t* zbf = (bf16_t*)(ws + 104 * MB);

  // 1. pack weights into bf16 WMMA fragment order
  pack_w_kernel<<<(cE * cE) / 256, 256, 0, stream>>>(Wq, Wqp, cE, cE);
  pack_w_kernel<<<(cE * cE) / 256, 256, 0, stream>>>(Wk, Wkp, cE, cE);
  pack_w_kernel<<<(cE * cE) / 256, 256, 0, stream>>>(Wv, Wvp, cE, cE);
  pack_w_kernel<<<(cE * cE) / 256, 256, 0, stream>>>(Wo, Wop, cE, cE);
  pack_w_kernel<<<(cE * cFF) / 256, 256, 0, stream>>>(W1, W1p, cE, cFF);
  pack_w_kernel<<<(cFF * cE) / 256, 256, 0, stream>>>(W2, W2p, cFF, cE);

  // 2. LN1
  ln_kernel<<<cM, 256, 0, stream>>>(x, g1, beta1, ybf);

  // 3. QKV projections (128x128 tiles)
  dim3 gEE(cE / 128, cM / 128);
  gemm_wmma_kernel<<<gEE, 128, 0, stream>>>(ybf, Wqp, bq, nullptr, qf, cM, cE, cE, 0);
  gemm_wmma_kernel<<<gEE, 128, 0, stream>>>(ybf, Wkp, bk, nullptr, kf, cM, cE, cE, 0);
  gemm_wmma_kernel<<<gEE, 128, 0, stream>>>(ybf, Wvp, bv, nullptr, vf, cM, cE, cE, 0);

  // 4. RoPE + per-head bf16 repack (V transposed)
  rope_pack_kernel<<<(cB * cH * cT * cD) / 256, 256, 0, stream>>>(qf, kf, vf, qh, kh, vt);

  // 5. causal flash attention (32 queries per wave)
  dim3 gAtt(cT / 32, cB * cH);
  attn_kernel<<<gAtt, 32, 0, stream>>>(qh, kh, vt, zbf);

  // 6. output projection + residual -> x1
  gemm_wmma_kernel<<<gEE, 128, 0, stream>>>(zbf, Wop, bo, x, x1, cM, cE, cE, 0);

  // 7. LN2
  ln_kernel<<<cM, 256, 0, stream>>>(x1, g2, beta2, ybf);

  // 8. FF1 with ReLU, bf16 output
  dim3 gFF1(cFF / 128, cM / 128);
  gemm_wmma_kernel<<<gFF1, 128, 0, stream>>>(ybf, W1p, b1, nullptr, hid, cM, cFF, cE,
                                             FLAG_RELU | FLAG_OUTBF16);

  // 9. FF2 + residual -> d_out (f32)
  gemm_wmma_kernel<<<gEE, 128, 0, stream>>>(hid, W2p, b2, x1, d_out, cM, cE, cFF, 0);
}
